// GraphedDenoiseLoop_17729624998123
// MI455X (gfx1250) — matrix operand order repaired and checked by reading, not compiled, not run on hardware
//
#include <hip/hip_runtime.h>
#include <hip/hip_bf16.h>
#include <stdint.h>

// ---------------------------------------------------------------------------
// MI455X (gfx1250) implementation of the 10-step denoise loop.
// Weight-bandwidth-bound => once per launch, convert weights to bf16 AND
// pre-tile them into the WMMA-B-fragment-native layout B'[K/32][N][32]
// (32 K-contiguous values per output column). The GEMM K-loop then moves both
// the A tile and the B tile with global_load_async_to_lds_b128 DMAs into
// double-buffered LDS (no in-loop transpose), loads fragments as aligned
// ds_load_b128 pairs, and issues 4 back-to-back v_wmma_f32_16x16x32_bf16.
// ---------------------------------------------------------------------------

typedef __attribute__((ext_vector_type(8)))  __bf16 bf16x8;
typedef __attribute__((ext_vector_type(16))) __bf16 bf16x16;
typedef __attribute__((ext_vector_type(8)))  float  f32x8;

namespace {
constexpr int CL  = 18;    // layers
constexpr int CD  = 1024;  // model dim
constexpr int CH  = 8;     // heads
constexpr int CDH = 256;   // head dim
constexpr int CFF = 4096;  // ffn dim
constexpr int CT  = 50;    // action tokens
constexpr int CS  = 51;    // seq (state + actions)
constexpr int CP  = 768;   // prefix len
constexpr int CM  = 64;    // padded rows (S -> 64)
constexpr int CTP = 896;   // padded kv length (7*128)
constexpr int CPT = CP + CS; // 819 valid kv entries
constexpr int CHD = CH * CDH; // 2048
}

// ======================= generic M=64 WMMA GEMM ============================
// C[64,N] = A[64,K](bf16, row-major) * B'(bf16, tiled [K/32][N][32]).
// f32 accumulate. N%128==0, K%32==0. Batched over blockIdx.z via sAz/sBz/sCz.
// Double-buffered: all tile traffic is async DMA; one barrier per K-step.
__global__ __launch_bounds__(256)
void k_gemm64(const __hip_bfloat16* __restrict__ A, long sAz, int lda,
              const __hip_bfloat16* __restrict__ Bt, long sBz, int bN,
              float* __restrict__ C, long sCz, int ldc, int K)
{
  __shared__ __hip_bfloat16 ldsA[2][CM * 32];    // 2 x 4 KB  A tiles
  __shared__ __hip_bfloat16 ldsB[2][128 * 32];   // 2 x 8 KB  B tiles (frag-native)

  A  += (long)blockIdx.z * sAz;
  Bt += (long)blockIdx.z * sBz;
  C  += (long)blockIdx.z * sCz;

  const int n0   = blockIdx.x * 128;
  const int tid  = threadIdx.x;
  const int lane = tid & 31;
  const int wid  = tid >> 5;      // 8 waves
  const int wm   = wid & 3;       // 16-row band of M
  const int wn   = wid >> 2;      // 64-col half of the N tile
  const int hi   = lane >> 4;     // lane-half (K select per ISA layout)
  const int l16  = lane & 15;

  // A staging: 256 threads x 16B cover the 64x32 tile
  const int arow = tid >> 2;            // 0..63
  const int ac8  = (tid & 3) * 8;       // 0,8,16,24 (bf16 elems)

  // low 32 bits of a generic LDS pointer == LDS byte offset (aperture encoding)
  const unsigned offA[2] = {
      (unsigned)(unsigned long long)(uintptr_t)&ldsA[0][arow * 32 + ac8],
      (unsigned)(unsigned long long)(uintptr_t)&ldsA[1][arow * 32 + ac8]};
  const unsigned offB[2] = {
      (unsigned)(unsigned long long)(uintptr_t)&ldsB[0][tid * 8],
      (unsigned)(unsigned long long)(uintptr_t)&ldsB[1][tid * 8]};

  f32x8 zv;
#pragma unroll
  for (int i = 0; i < 8; ++i) zv[i] = 0.f;
  f32x8 acc[4] = {zv, zv, zv, zv};

  auto issueA = [&](int kt, int buf) {
    unsigned long long ga =
        (unsigned long long)(uintptr_t)(A + (long)arow * lda + kt + ac8);
    asm volatile("global_load_async_to_lds_b128 %0, %1, off"
                 :: "v"(offA[buf]), "v"(ga) : "memory");
  };
  // B tile is a contiguous 8KB block in the pre-tiled layout: pure DMA copy.
  auto issueB = [&](int kt, int buf) {
    unsigned long long ga = (unsigned long long)(uintptr_t)
        (Bt + ((long)(kt >> 5) * bN + n0) * 32 + tid * 8);
    asm volatile("global_load_async_to_lds_b128 %0, %1, off"
                 :: "v"(offB[buf]), "v"(ga) : "memory");
    asm volatile("global_load_async_to_lds_b128 %0, %1, off"
                 :: "v"(offB[buf] + 4096u), "v"(ga + 4096ull) : "memory");
  };

  const int nk = K >> 5;

  // ---- prologue: stage tile 0 into buffer 0
  issueA(0, 0);
  issueB(0, 0);
  asm volatile("s_wait_asynccnt 0x0" ::: "memory");
  __syncthreads();

  int cur = 0;
  for (int i = 0; i < nk; ++i) {
    const int  kt       = i << 5;
    const bool has_next = (i + 1 < nk);

    // ---- issue next tile's DMAs early (they land while we do WMMA below)
    if (has_next) {
      issueA(kt + 32, cur ^ 1);
      issueB(kt + 32, cur ^ 1);
      if (i + 2 < nk)
        __builtin_prefetch(
            (const void*)(Bt + ((long)((kt + 64) >> 5) * bN + n0) * 32 + tid * 8),
            0, 1);
    }

    // ---- compute tile i from buffer `cur`
    // A fragment per documented 16-bit A 16x32 layout:
    // lanes 0-15: K 0..7 then 16..23 ; lanes 16-31: K 8..15 then 24..31
    const __bf16* la = (const __bf16*)ldsA[cur] + (wm * 16 + l16) * 32 + hi * 8;
    bf16x8 a0 = *(const bf16x8*)(la);
    bf16x8 a1 = *(const bf16x8*)(la + 16);
    bf16x16 afrag = __builtin_shufflevector(a0, a1,
        0,1,2,3,4,5,6,7,8,9,10,11,12,13,14,15);

    // load all four B fragments, then issue the WMMAs back-to-back
    bf16x16 bfrag[4];
#pragma unroll
    for (int j = 0; j < 4; ++j) {
      // frag-native LDS layout [n][32]: lanes 0-15 get K 0..7,16..23; lanes 16-31 K 8..15,24..31
      const __bf16* lb =
          (const __bf16*)ldsB[cur] + (wn * 64 + j * 16 + l16) * 32 + hi * 8;
      bf16x8 b0 = *(const bf16x8*)(lb);
      bf16x8 b1 = *(const bf16x8*)(lb + 16);
      bfrag[j] = __builtin_shufflevector(b0, b1,
          0,1,2,3,4,5,6,7,8,9,10,11,12,13,14,15);
    }
#if __has_builtin(__builtin_amdgcn_sched_barrier)
    __builtin_amdgcn_sched_barrier(0);   // keep loads grouped ahead of the WMMAs
#endif
#pragma unroll
    for (int j = 0; j < 4; ++j)
      acc[j] = __builtin_amdgcn_wmma_f32_16x16x32_bf16(
          false, afrag, false, bfrag[j], (short)0, acc[j], false, false);

    asm volatile("s_wait_asynccnt 0x0" ::: "memory");
    __syncthreads();
    cur ^= 1;
  }

  // C layout: VGPR r, lanes 0-15 -> M=r, lanes 16-31 -> M=r+8; N = lane&15
#pragma unroll
  for (int j = 0; j < 4; ++j)
#pragma unroll
    for (int r = 0; r < 8; ++r) {
      int m = wm * 16 + r + 8 * hi;
      int n = n0 + wn * 64 + j * 16 + l16;
      C[(long)m * ldc + n] = acc[j][r];
    }
}

// ============================ utility kernels ==============================
// One-time convert+tile: f32 row-major W[KR x N] -> bf16 B'[KR/32][N][32].
// Block transposes a 32(k) x 64(n) tile through LDS; reads and writes coalesced.
__global__ __launch_bounds__(256)
void k_cvtT(const float* __restrict__ W, __hip_bfloat16* __restrict__ Bt, int N) {
  __shared__ float tile[32][65];             // +1 pad: conflict-free column reads
  const int kt = blockIdx.y, n0 = blockIdx.x * 64, tid = threadIdx.x;
  const int r  = tid >> 3;                   // 0..31 (k row)
  const int c8 = (tid & 7) * 8;              // 0..56
  const float* src = W + ((long)kt * 32 + r) * N + n0 + c8;
#pragma unroll
  for (int e = 0; e < 8; ++e) tile[r][c8 + e] = src[e];
  __syncthreads();
  const int n  = tid >> 2;                   // 0..63
  const int k8 = (tid & 3) * 8;              // 0,8,16,24
  __hip_bfloat16* dst = Bt + ((long)kt * N + n0 + n) * 32 + k8;
#pragma unroll
  for (int e = 0; e < 8; ++e) dst[e] = __float2bfloat16(tile[k8 + e][n]);
}

__global__ void k_cvt_bf16(const float* __restrict__ s,
                           __hip_bfloat16* __restrict__ d, long n) {
  long i = (long)blockIdx.x * blockDim.x + threadIdx.x;
  long st = (long)gridDim.x * blockDim.x;
  for (; i < n; i += st) d[i] = __float2bfloat16(s[i]);
}

__global__ void k_zero_u16(unsigned short* __restrict__ p, long n) {
  long i = (long)blockIdx.x * blockDim.x + threadIdx.x;
  long st = (long)gridDim.x * blockDim.x;
  for (; i < n; i += st) p[i] = 0;
}

__global__ void k_copy_f32(const float* __restrict__ s, float* __restrict__ d, long n) {
  long i = (long)blockIdx.x * blockDim.x + threadIdx.x;
  long st = (long)gridDim.x * blockDim.x;
  for (; i < n; i += st) d[i] = s[i];
}

// prefix K / V caches in bf16, stored in the GEMM-B tiled layout.
// pk/pv layout: [L][1][H][P][DH].
// kT (scores B, K-dim=DH, N=CTP): per head [DH/32][CTP][32], elem (d,t).
// vC (att    B, K-dim=CTP, N=DH): per head [CTP/32][DH][32], elem (t,d).
__global__ void k_build_kvT(const float* __restrict__ pk, const float* __restrict__ pv,
                            __hip_bfloat16* __restrict__ kT, __hip_bfloat16* __restrict__ vC) {
  const long n = (long)CL * CH * CP * CDH;
  long i = (long)blockIdx.x * blockDim.x + threadIdx.x;
  long st = (long)gridDim.x * blockDim.x;
  for (; i < n; i += st) {
    long d  = i & (CDH - 1);
    long t  = (i >> 8) % CP;
    long lh = i / ((long)CP * CDH);      // l*H + h
    kT[lh * CDH * CTP + (d >> 5) * (CTP * 32) + t * 32 + (d & 31)] =
        __float2bfloat16(pk[i]);
    vC[lh * CTP * CDH + (t >> 5) * (CDH * 32) + d * 32 + (t & 31)] =
        __float2bfloat16(pv[i]);
  }
}

__global__ void k_offset(const unsigned char* __restrict__ m, float* __restrict__ off) {
  if (blockIdx.x == 0 && threadIdx.x == 0) {
    int c = 0;
    for (int i = 0; i < CP; ++i) c += (m[i] != 0);
    off[0] = (float)c;
  }
}

__global__ void k_time_embed(float tval, float* __restrict__ te) {
  int i = blockIdx.x * 256 + threadIdx.x;
  if (i < CD / 2) {
    float frac   = (float)i * (1.f / 511.f);
    float period = 0.004f * exp2f(frac * 9.965784284662087f);   // 1000^frac
    float ang    = tval / period * 6.283185307179586f;
    te[i]           = sinf(ang);
    te[CD / 2 + i]  = cosf(ang);
  }
}

// y[n] = act(sum_k x[k] * W[k*N+n]);  act: 0=none, 1=silu
__global__ void k_gemv(const float* __restrict__ x, const float* __restrict__ W,
                       float* __restrict__ y, int K, int N, int act) {
  int n = blockIdx.x * blockDim.x + threadIdx.x;
  if (n >= N) return;
  float s = 0.f;
  for (int k = 0; k < K; ++k) s += x[k] * W[(long)k * N + n];
  if (act == 1) s = s / (1.f + expf(-s));
  y[n] = s;
}

// h[0]=state@W_state ; h[1..50]=x_t@W_action_in ; pad rows zero
__global__ void k_embed(const float* __restrict__ state, const float* __restrict__ xt,
                        const float* __restrict__ Wst, const float* __restrict__ Wain,
                        float* __restrict__ h) {
  const int s = blockIdx.x, tid = threadIdx.x;
  for (int c = tid; c < CD; c += 256) {
    float acc = 0.f;
    if (s == 0)       for (int k = 0; k < 32; ++k) acc += state[k] * Wst[k * CD + c];
    else if (s <= CT) for (int k = 0; k < 32; ++k) acc += xt[(s - 1) * 32 + k] * Wain[k * CD + c];
    h[(long)s * CD + c] = acc;
  }
}

// adaptive RMSNorm -> bf16 (scale = ada[0..D), gate consumed later)
__global__ __launch_bounds__(256)
void k_adarms(const float* __restrict__ h, const float* __restrict__ wn,
              const float* __restrict__ ada, __hip_bfloat16* __restrict__ out) {
  const int r = blockIdx.x, tid = threadIdx.x;
  const float* row = h + (long)r * CD;
  __shared__ float red[256];
  float s = 0.f;
  for (int c = tid; c < CD; c += 256) { float v = row[c]; s += v * v; }
  red[tid] = s; __syncthreads();
  for (int o = 128; o > 0; o >>= 1) { if (tid < o) red[tid] += red[tid + o]; __syncthreads(); }
  const float rs = rsqrtf(red[0] * (1.f / CD) + 1e-6f);
  for (int c = tid; c < CD; c += 256)
    out[(long)r * CD + c] = __float2bfloat16(row[c] * rs * (1.f + wn[c]) * (1.f + ada[c]));
}

// final RMSNorm -> f32
__global__ __launch_bounds__(256)
void k_finalnorm(const float* __restrict__ h, const float* __restrict__ wn,
                 const float* __restrict__ ada, float* __restrict__ out) {
  const int r = blockIdx.x, tid = threadIdx.x;
  const float* row = h + (long)r * CD;
  __shared__ float red[256];
  float s = 0.f;
  for (int c = tid; c < CD; c += 256) { float v = row[c]; s += v * v; }
  red[tid] = s; __syncthreads();
  for (int o = 128; o > 0; o >>= 1) { if (tid < o) red[tid] += red[tid + o]; __syncthreads(); }
  const float rs = rsqrtf(red[0] * (1.f / CD) + 1e-6f);
  for (int c = tid; c < CD; c += 256)
    out[(long)r * CD + c] = row[c] * rs * (1.f + wn[c]) * (1.f + ada[c]);
}

// RoPE; writes q (scaled by 1/sqrt(DH)) into [H][64][DH] bf16 (A-side),
// k into this layer's tiled K cache at t=768+s, v into tiled V cache.
__global__ void k_rope_qkv(const float* __restrict__ qf, const float* __restrict__ kf,
                           const float* __restrict__ vf, const float* __restrict__ offp,
                           __hip_bfloat16* __restrict__ qA, __hip_bfloat16* __restrict__ kT,
                           __hip_bfloat16* __restrict__ vC) {
  const int h = blockIdx.x, s = blockIdx.y, d = threadIdx.x;   // d in [0,256)
  if (s >= CS) { qA[((long)h * CM + s) * CDH + d] = __float2bfloat16(0.f); return; }
  const float pos  = offp[0] + (float)s;
  const int   i    = d & 127;
  const float invf = exp2f(-(float)(2 * i) * (13.287712379549449f / 256.f)); // 10000^(-2i/256)
  const float ang  = pos * invf;
  const float c = cosf(ang), sn = sinf(ang);
  const int base = s * CHD + h * CDH;
  float xq = qf[base + d];
  float rq = (d < 128) ? -qf[base + d + 128] : qf[base + d - 128];
  float xk = kf[base + d];
  float rk = (d < 128) ? -kf[base + d + 128] : kf[base + d - 128];
  qA[((long)h * CM + s) * CDH + d] = __float2bfloat16((xq * c + rq * sn) * 0.0625f);
  const int t = CP + s;
  kT[(long)h * CDH * CTP + (d >> 5) * (CTP * 32) + t * 32 + (d & 31)] =
      __float2bfloat16(xk * c + rk * sn);
  vC[(long)h * CTP * CDH + (t >> 5) * (CDH * 32) + d * 32 + (t & 31)] =
      __float2bfloat16(vf[base + d]);
}

__global__ __launch_bounds__(256)
void k_softmax(const float* __restrict__ scores, __hip_bfloat16* __restrict__ probs) {
  const int h = blockIdx.x, s = blockIdx.y, tid = threadIdx.x;
  const float* row = scores + ((long)h * CM + s) * CTP;
  __hip_bfloat16* prow = probs + ((long)h * CM + s) * CTP;
  if (s >= CS) { for (int t = tid; t < CTP; t += 256) prow[t] = __float2bfloat16(0.f); return; }
  __shared__ float red[256];
  float mx = -1e30f;
  for (int t = tid; t < CPT; t += 256) mx = fmaxf(mx, row[t]);
  red[tid] = mx; __syncthreads();
  for (int o = 128; o > 0; o >>= 1) { if (tid < o) red[tid] = fmaxf(red[tid], red[tid + o]); __syncthreads(); }
  mx = red[0]; __syncthreads();
  float sm = 0.f;
  for (int t = tid; t < CPT; t += 256) sm += expf(row[t] - mx);
  red[tid] = sm; __syncthreads();
  for (int o = 128; o > 0; o >>= 1) { if (tid < o) red[tid] += red[tid + o]; __syncthreads(); }
  const float inv = 1.f / red[0];
  for (int t = tid; t < CTP; t += 256)
    prow[t] = __float2bfloat16(t < CPT ? expf(row[t] - mx) * inv : 0.f);
}

__global__ void k_gelu_mul(const float* __restrict__ g, const float* __restrict__ u,
                           __hip_bfloat16* __restrict__ o, long n) {
  long i = (long)blockIdx.x * blockDim.x + threadIdx.x;
  long st = (long)gridDim.x * blockDim.x;
  for (; i < n; i += st) {
    float x  = g[i];
    float tt = 0.7978845608028654f * (x + 0.044715f * x * x * x);
    o[i] = __float2bfloat16(0.5f * x * (1.f + tanhf(tt)) * u[i]);
  }
}

__global__ void k_accum_gated(float* __restrict__ h, const float* __restrict__ src,
                              const float* __restrict__ gate, long n) {
  long i = (long)blockIdx.x * blockDim.x + threadIdx.x;
  long st = (long)gridDim.x * blockDim.x;
  for (; i < n; i += st) h[i] += src[i] * gate[i & (CD - 1)];
}

// x_t[s] += dt * (hn[s+1] @ W_action_out)   (block = one wave32, thread = action dim)
__global__ void k_actout(const float* __restrict__ hn, const float* __restrict__ Wao,
                         float* __restrict__ xt, float dt) {
  const int s = blockIdx.x, a = threadIdx.x;
  float acc = 0.f;
  for (int k = 0; k < CD; ++k) acc += hn[(long)(s + 1) * CD + k] * Wao[k * 32 + a];
  xt[s * 32 + a] += dt * acc;
}

// ============================== host side ==================================
extern "C" void kernel_launch(void* const* d_in, const int* in_sizes, int n_in,
                              void* d_out, int out_size, void* d_ws, size_t ws_size,
                              hipStream_t stream) {
  (void)in_sizes; (void)n_in; (void)out_size; (void)ws_size;

  const float* state    = (const float*)d_in[0];
  const float* noise    = (const float*)d_in[1];
  const float* prefix_k = (const float*)d_in[2];
  const float* prefix_v = (const float*)d_in[3];
  const float* Wain     = (const float*)d_in[4];
  const float* Wst      = (const float*)d_in[5];
  const float* Wq       = (const float*)d_in[6];
  const float* Wk       = (const float*)d_in[7];
  const float* Wv       = (const float*)d_in[8];
  const float* Wo       = (const float*)d_in[9];
  const float* Wa1      = (const float*)d_in[10];
  const float* Wa2      = (const float*)d_in[11];
  const float* Waf      = (const float*)d_in[12];
  const float* wn1      = (const float*)d_in[13];
  const float* wn2      = (const float*)d_in[14];
  const float* wnf      = (const float*)d_in[15];
  const float* Wg       = (const float*)d_in[16];
  const float* Wu       = (const float*)d_in[17];
  const float* Wd       = (const float*)d_in[18];
  const float* Wti      = (const float*)d_in[19];
  const float* Wto      = (const float*)d_in[20];
  const float* Wao      = (const float*)d_in[21];
  const unsigned char* pmask = (const unsigned char*)d_in[22];

  // ---- bump allocator over workspace
  uintptr_t cur = (uintptr_t)d_ws;
  auto alloc = [&](size_t bytes) -> void* {
    cur = (cur + 255) & ~(uintptr_t)255;
    void* p = (void*)cur; cur += bytes; return p;
  };

  const long nQ  = (long)CL * CD * CHD;   // q/k/v/o weight elems per tensor
  const long nF  = (long)CL * CD * CFF;   // gate/up/down weight elems
  const long nKT = (long)CL * CH * CDH * CTP;
  const long nVC = (long)CL * CH * CTP * CDH;

  __hip_bfloat16* wq  = (__hip_bfloat16*)alloc(nQ * 2);
  __hip_bfloat16* wk  = (__hip_bfloat16*)alloc(nQ * 2);
  __hip_bfloat16* wv  = (__hip_bfloat16*)alloc(nQ * 2);
  __hip_bfloat16* wo  = (__hip_bfloat16*)alloc(nQ * 2);
  __hip_bfloat16* wg  = (__hip_bfloat16*)alloc(nF * 2);
  __hip_bfloat16* wu  = (__hip_bfloat16*)alloc(nF * 2);
  __hip_bfloat16* wd  = (__hip_bfloat16*)alloc(nF * 2);
  __hip_bfloat16* kT  = (__hip_bfloat16*)alloc(nKT * 2);
  __hip_bfloat16* vC  = (__hip_bfloat16*)alloc(nVC * 2);

  float* xt    = (float*)alloc(CT * 32 * 4);
  float* off   = (float*)alloc(4);
  float* te    = (float*)alloc(CD * 4);
  float* cond1 = (float*)alloc(CD * 4);
  float* cond  = (float*)alloc(CD * 4);
  float* hbuf  = (float*)alloc((long)CM * CD * 4);
  float* ada1  = (float*)alloc(2 * CD * 4);
  float* ada2  = (float*)alloc(2 * CD * 4);
  float* adaf  = (float*)alloc(2 * CD * 4);
  __hip_bfloat16* nbf = (__hip_bfloat16*)alloc((long)CM * CD * 2);
  float* qf    = (float*)alloc((long)CM * CHD * 4);
  float* kf    = (float*)alloc((long)CM * CHD * 4);
  float* vf    = (float*)alloc((long)CM * CHD * 4);
  __hip_bfloat16* qA  = (__hip_bfloat16*)alloc((long)CH * CM * CDH * 2);
  float* scores = (float*)alloc((long)CH * CM * CTP * 4);
  __hip_bfloat16* probs = (__hip_bfloat16*)alloc((long)CH * CM * CTP * 2);
  float* attp   = (float*)alloc((long)CM * CHD * 4);
  __hip_bfloat16* attp_bf = (__hip_bfloat16*)alloc((long)CM * CHD * 2);
  float* attO   = (float*)alloc((long)CM * CD * 4);
  float* t1     = (float*)alloc((long)CM * CFF * 4);
  float* t2     = (float*)alloc((long)CM * CFF * 4);
  __hip_bfloat16* mbf = (__hip_bfloat16*)alloc((long)CM * CFF * 2);
  float* mlpO   = (float*)alloc((long)CM * CD * 4);
  float* hn     = (float*)alloc((long)CM * CD * 4);

  auto gsz = [](long n) -> int {
    long b = (n + 255) / 256; return (int)(b > 8192 ? 8192 : b);
  };
  auto gemm = [&](const __hip_bfloat16* A, long sAz, int lda,
                  const __hip_bfloat16* B, long sBz, int bN,
                  float* C, long sCz, int ldc, int K, int N, int Z) {
    dim3 g((unsigned)(N / 128), 1u, (unsigned)Z);
    k_gemm64<<<g, 256, 0, stream>>>(A, sAz, lda, B, sBz, bN, C, sCz, ldc, K);
  };
  // convert+tile a row-major f32 weight [KR x N] into B'[KR/32][N][32] bf16
  auto cvtT = [&](const float* W, __hip_bfloat16* Bt, long KR, int N) {
    dim3 g((unsigned)(N / 64), (unsigned)(KR / 32));
    k_cvtT<<<g, 256, 0, stream>>>(W, Bt, N);
  };

  // ---- per-launch init: tiled bf16 weights, kv caches, x_t, rope offset
  cvtT(Wq, wq, (long)CL * CD, CHD);
  cvtT(Wk, wk, (long)CL * CD, CHD);
  cvtT(Wv, wv, (long)CL * CD, CHD);
  cvtT(Wo, wo, (long)CL * CHD, CD);
  cvtT(Wg, wg, (long)CL * CD, CFF);
  cvtT(Wu, wu, (long)CL * CD, CFF);
  cvtT(Wd, wd, (long)CL * CFF, CD);
  k_zero_u16<<<gsz(nKT), 256, 0, stream>>>((unsigned short*)kT, nKT);
  k_zero_u16<<<gsz(nVC), 256, 0, stream>>>((unsigned short*)vC, nVC);
  k_build_kvT<<<gsz((long)CL * CH * CP * CDH), 256, 0, stream>>>(prefix_k, prefix_v, kT, vC);
  k_copy_f32<<<7, 256, 0, stream>>>(noise, xt, CT * 32);
  k_offset<<<1, 64, 0, stream>>>(pmask, off);

  const float dt = -1.f / 10.f;
  for (int step = 0; step < 10; ++step) {
    const float tval = 1.f + step * dt;
    k_time_embed<<<2, 256, 0, stream>>>(tval, te);
    k_gemv<<<4, 256, 0, stream>>>(te, Wti, cond1, CD, CD, 1);
    k_gemv<<<4, 256, 0, stream>>>(cond1, Wto, cond, CD, CD, 1);
    k_embed<<<CM, 256, 0, stream>>>(state, xt, Wst, Wain, hbuf);

    for (int l = 0; l < CL; ++l) {
      __hip_bfloat16* kTl = kT + (long)l * CH * CDH * CTP;
      __hip_bfloat16* vCl = vC + (long)l * CH * CTP * CDH;

      // --- attention block
      k_gemv<<<8, 256, 0, stream>>>(cond, Wa1 + (long)l * CD * 2 * CD, ada1, CD, 2 * CD, 0);
      k_adarms<<<CM, 256, 0, stream>>>(hbuf, wn1 + (long)l * CD, ada1, nbf);
      gemm(nbf, 0, CD, wq + (long)l * CD * CHD, 0, CHD, qf, 0, CHD, CD, CHD, 1);
      gemm(nbf, 0, CD, wk + (long)l * CD * CHD, 0, CHD, kf, 0, CHD, CD, CHD, 1);
      gemm(nbf, 0, CD, wv + (long)l * CD * CHD, 0, CHD, vf, 0, CHD, CD, CHD, 1);
      k_rope_qkv<<<dim3(CH, CM), CDH, 0, stream>>>(qf, kf, vf, off, qA, kTl, vCl);
      // scores[h] = q[h] @ K[h]^T   (tiled K cache, K-dim = DH)
      gemm(qA, (long)CM * CDH, CDH, kTl, (long)CDH * CTP, CTP,
           scores, (long)CM * CTP, CTP, CDH, CTP, CH);
      k_softmax<<<dim3(CH, CM), 256, 0, stream>>>(scores, probs);
      // att[h] = P[h] @ V[h]  (tiled V cache, K-dim = CTP), packed via ldc/sCz
      gemm(probs, (long)CM * CTP, CTP, vCl, (long)CTP * CDH, CDH,
           attp, CDH, CHD, CTP, CDH, CH);
      k_cvt_bf16<<<gsz((long)CM * CHD), 256, 0, stream>>>(attp, attp_bf, (long)CM * CHD);
      gemm(attp_bf, 0, CHD, wo + (long)l * CHD * CD, 0, CD, attO, 0, CD, CHD, CD, 1);
      k_accum_gated<<<gsz((long)CM * CD), 256, 0, stream>>>(hbuf, attO, ada1 + CD, (long)CM * CD);

      // --- MLP block
      k_gemv<<<8, 256, 0, stream>>>(cond, Wa2 + (long)l * CD * 2 * CD, ada2, CD, 2 * CD, 0);
      k_adarms<<<CM, 256, 0, stream>>>(hbuf, wn2 + (long)l * CD, ada2, nbf);
      gemm(nbf, 0, CD, wg + (long)l * CD * CFF, 0, CFF, t1, 0, CFF, CD, CFF, 1);
      gemm(nbf, 0, CD, wu + (long)l * CD * CFF, 0, CFF, t2, 0, CFF, CD, CFF, 1);
      k_gelu_mul<<<gsz((long)CM * CFF), 256, 0, stream>>>(t1, t2, mbf, (long)CM * CFF);
      gemm(mbf, 0, CFF, wd + (long)l * CFF * CD, 0, CD, mlpO, 0, CD, CFF, CD, 1);
      k_accum_gated<<<gsz((long)CM * CD), 256, 0, stream>>>(hbuf, mlpO, ada2 + CD, (long)CM * CD);
    }

    k_gemv<<<8, 256, 0, stream>>>(cond, Waf, adaf, CD, 2 * CD, 0);
    k_finalnorm<<<CM, 256, 0, stream>>>(hbuf, wnf, adaf, hn);
    k_actout<<<CT, 32, 0, stream>>>(hn, Wao, xt, dt);
  }

  k_copy_f32<<<7, 256, 0, stream>>>(xt, (float*)d_out, CT * 32);
}
